// MatchLSTM_81080392614914
// MI455X (gfx1250) — matrix operand order repaired and checked by reading, not compiled, not run on hardware
//
#include <hip/hip_runtime.h>
#include <math.h>

// ---------------------------------------------------------------------------
// CDNA5 (gfx1250) wave32 WMMA helpers: f16 inputs, f32 accumulate, 16x16x32.
// ---------------------------------------------------------------------------
typedef __attribute__((ext_vector_type(16))) _Float16 v16h;
typedef __attribute__((ext_vector_type(8)))  _Float16 v8h;
typedef __attribute__((ext_vector_type(8)))  float    v8f;
typedef __attribute__((ext_vector_type(4)))  int      v4i_t;
typedef __attribute__((ext_vector_type(8)))  int      v8i_t;
typedef __attribute__((ext_vector_type(4)))  unsigned int v4u_t;

#define DEV static __device__ __forceinline__

// Optional CDNA5 async global->LDS staging (ASYNCcnt path). Guarded so the
// build cleanly falls back to plain b128 copies if the builtins are absent.
#if defined(__HIP_DEVICE_COMPILE__) && \
    __has_builtin(__builtin_amdgcn_global_load_async_to_lds_b128) && \
    __has_builtin(__builtin_amdgcn_s_wait_asynccnt)
#define HAVE_ASYNC_LDS 1
#else
#define HAVE_ASYNC_LDS 0
#endif

DEV v8f wmma32(v16h a, v16h b, v8f c) {
  // (neg_a, A, neg_b, B, c_mod, C, reuse_a, reuse_b)
  return __builtin_amdgcn_wmma_f32_16x16x32_f16(false, a, false, b, (short)0, c,
                                                false, false);
}
// ISA 7.12.2 16-bit A 16x32: lane half j -> K = (j&7) + ((j>=8)?16:0) + hi*8
DEV int koffA(int j, int hi) { return (j & 7) + ((j >> 3) << 4) + hi * 8; }

DEV v16h fragA(const _Float16* base, int stride, int m, int kBase, int lane) {
  int mm = lane & 15, hi = lane >> 4;
  v16h a;
#pragma unroll
  for (int j = 0; j < 16; ++j) a[j] = base[(m + mm) * stride + kBase + hi * 16 + j];
  return a;
}
// B fragment where logical B[k][n] = W[n][k] (W row-major [N][K])
DEV v16h fragB_nt(const _Float16* w, int stride, int n, int kBase, int lane) {
  int nn = lane & 15, hi = lane >> 4;
  v16h b;
#pragma unroll
  for (int j = 0; j < 16; ++j) b[j] = w[(n + nn) * stride + kBase + hi * 16 + j];
  return b;
}
// B fragment from row-major X[K][N]
DEV v16h fragB_nn(const _Float16* x, int stride, int kBase, int n, int lane) {
  int nn = lane & 15, hi = lane >> 4;
  v16h b;
#pragma unroll
  for (int j = 0; j < 16; ++j) b[j] = x[(kBase + hi * 16 + j) * stride + n + nn];
  return b;
}
DEV v8h load8h(const _Float16* p) {  // unaligned-tolerant 16B load
  v8h v;
  __builtin_memcpy(&v, p, 16);
  return v;
}
// 16-byte global -> LDS copy; async path when available + aligned.
DEV void g2l16(_Float16* dst, const _Float16* src, bool useAsync) {
  (void)useAsync;
#if HAVE_ASYNC_LDS
  if (useAsync) {
    __builtin_amdgcn_global_load_async_to_lds_b128(
        (__attribute__((address_space(1))) v4i_t*)src,
        (__attribute__((address_space(3))) v4i_t*)dst, 0, 0);
    return;
  }
#endif
  *(v8h*)dst = load8h(src);
}
DEV void stage_wait() {
#if HAVE_ASYNC_LDS
  __builtin_amdgcn_s_wait_asynccnt(0);
#endif
}
DEV float sigf(float x) { return 1.0f / (1.0f + __expf(-x)); }
DEV float actf(float v, int act) {
  if (act == 1) return fmaxf(v, 0.0f);
  if (act == 2) return sigf(v);
  return v;
}

// ---------------------------------------------------------------------------
// GEMM (f16 operands, f32 accumulate): C[M,N] = act(A[M,K] @ W[N,K]^T + bias)
// 256 threads = 8 waves, tile 32x64, K-step 64 (2 WMMAs per barrier).
// ---------------------------------------------------------------------------
__global__ __launch_bounds__(256) void k_gemm_nt(
    const _Float16* __restrict__ A, const _Float16* __restrict__ W,
    const float* __restrict__ bias, float* __restrict__ C,
    int M, int N, int K, int act) {
  __shared__ _Float16 As[32 * 64];
  __shared__ _Float16 Ws[64 * 64];
  const int tid = threadIdx.x, lane = tid & 31, w = tid >> 5;
  const int mt = w & 1, wn = w >> 1;
  const int mBase = blockIdx.y * 32, nBase = blockIdx.x * 64;
  const bool alignedK = ((K & 7) == 0);
  v8f c = {};
  for (int kt = 0; kt < K; kt += 64) {
    const bool fullK = (kt + 64 <= K);
    {  // A tile 32x64: one 8-half chunk per thread
      int r = tid >> 3, col = (tid & 7) * 8;
      int gm = mBase + r, gk = kt + col;
      _Float16* dst = &As[r * 64 + col];
      if (fullK && gm < M) {
        g2l16(dst, &A[(size_t)gm * K + gk], alignedK);
      } else {
        for (int j = 0; j < 8; ++j)
          dst[j] = (gm < M && gk + j < K) ? A[(size_t)gm * K + gk + j] : (_Float16)0.0f;
      }
    }
    for (int e = tid; e < 512; e += 256) {  // W tile 64x64
      int r = e >> 3, col = (e & 7) * 8;
      int gn = nBase + r, gk = kt + col;
      _Float16* dst = &Ws[r * 64 + col];
      if (fullK && gn < N) {
        g2l16(dst, &W[(size_t)gn * K + gk], alignedK);
      } else {
        for (int j = 0; j < 8; ++j)
          dst[j] = (gn < N && gk + j < K) ? W[(size_t)gn * K + gk + j] : (_Float16)0.0f;
      }
    }
    stage_wait();
    __syncthreads();
    if (kt + 64 < K)  // speculative prefetch of next A tile
      __builtin_prefetch(&A[(size_t)(mBase + lane) * K + kt + 64], 0, 1);
    c = wmma32(fragA(As, 64, mt * 16, 0, lane), fragB_nt(Ws, 64, wn * 16, 0, lane), c);
    c = wmma32(fragA(As, 64, mt * 16, 32, lane), fragB_nt(Ws, 64, wn * 16, 32, lane), c);
    __syncthreads();
  }
  int hi = lane >> 4;
  int n = nBase + wn * 16 + (lane & 15);
#pragma unroll
  for (int r = 0; r < 8; ++r) {
    int m = mBase + mt * 16 + hi * 8 + r;
    if (m < M && n < N) {
      float v = c[r] + (bias ? bias[n] : 0.0f);
      C[(size_t)m * N + n] = actf(v, act);
    }
  }
}

// ---------------------------------------------------------------------------
// Batched channel-major GEMM: C[b][M,N] = act(W[M,K] @ X[b][K,N] + bias[M])
// ---------------------------------------------------------------------------
__global__ __launch_bounds__(256) void k_gemm_nn(
    const _Float16* __restrict__ Wm, const _Float16* __restrict__ X,
    const float* __restrict__ bias, float* __restrict__ C,
    int M, int N, int K, int act) {
  __shared__ _Float16 Wt[32 * 64];
  __shared__ _Float16 Xt[64 * 64];
  const int tid = threadIdx.x, lane = tid & 31, w = tid >> 5;
  const int mt = w & 1, wn = w >> 1;
  const int mBase = blockIdx.y * 32, nBase = blockIdx.x * 64;
  const int bb = blockIdx.z;
  const _Float16* Xb = X + (size_t)bb * K * N;
  float* Cb = C + (size_t)bb * M * N;
  const bool alignedK = ((K & 7) == 0);
  const bool alignedN = ((N & 7) == 0);
  v8f c = {};
  for (int kt = 0; kt < K; kt += 64) {
    const bool fullK = (kt + 64 <= K);
    {  // W tile 32x64 (m x k)
      int r = tid >> 3, col = (tid & 7) * 8;
      int gm = mBase + r, gk = kt + col;
      _Float16* dst = &Wt[r * 64 + col];
      if (fullK && gm < M) {
        g2l16(dst, &Wm[(size_t)gm * K + gk], alignedK);
      } else {
        for (int j = 0; j < 8; ++j)
          dst[j] = (gm < M && gk + j < K) ? Wm[(size_t)gm * K + gk + j] : (_Float16)0.0f;
      }
    }
    const bool fullN = (nBase + 64 <= N);
    for (int e = tid; e < 512; e += 256) {  // X tile 64x64 (k x n)
      int r = e >> 3, col = (e & 7) * 8;
      int gk = kt + r, gn = nBase + col;
      _Float16* dst = &Xt[r * 64 + col];
      if (fullN && gk < K) {
        g2l16(dst, &Xb[(size_t)gk * N + gn], alignedN);
      } else {
        for (int j = 0; j < 8; ++j)
          dst[j] = (gk < K && gn + j < N) ? Xb[(size_t)gk * N + gn + j] : (_Float16)0.0f;
      }
    }
    stage_wait();
    __syncthreads();
    c = wmma32(fragA(Wt, 64, mt * 16, 0, lane), fragB_nn(Xt, 64, 0, wn * 16, lane), c);
    c = wmma32(fragA(Wt, 64, mt * 16, 32, lane), fragB_nn(Xt, 64, 32, wn * 16, lane), c);
    __syncthreads();
  }
  int hi = lane >> 4;
  int n = nBase + wn * 16 + (lane & 15);
#pragma unroll
  for (int r = 0; r < 8; ++r) {
    int m = mBase + mt * 16 + hi * 8 + r;
    if (m < M && n < N) {
      float v = c[r] + (bias ? bias[m] : 0.0f);
      Cb[(size_t)m * N + n] = actf(v, act);
    }
  }
}

// ---------------------------------------------------------------------------
// TDM probe/warm: descriptor-driven 2D tile load (64x32 f16) into LDS, then
// s_wait_tensorcnt. 6-arg builtin form (clang-23/therock style).
// ---------------------------------------------------------------------------
__global__ void k_tdm_warm(const _Float16* __restrict__ src, float* __restrict__ sink) {
#if defined(__HIP_DEVICE_COMPILE__) && \
    __has_builtin(__builtin_amdgcn_tensor_load_to_lds) && \
    __has_builtin(__builtin_amdgcn_s_wait_tensorcnt)
  __shared__ _Float16 buf[64 * 32];
  unsigned long long ga = (unsigned long long)(size_t)src;
  v4u_t g0;
  g0.x = 1u;                                   // count=1 (valid descriptor)
  g0.y = 0u;                                   // lds_addr: buf is the only LDS var
  g0.z = (unsigned)(ga & 0xffffffffu);         // global_addr[31:0]
  g0.w = (unsigned)((ga >> 32) & 0x1ffffffu) | (2u << 30);  // addr[56:32] | type=2
  v8i_t g1;
  g1.s0 = (1 << 16);                           // data_size = 2 bytes
  g1.s1 = (int)((256u & 0xffffu) << 16);       // tensor_dim0 = 256 (lo16)
  g1.s2 = (int)((256u >> 16) | ((1024u & 0xffffu) << 16));  // dim0 hi | dim1 lo
  g1.s3 = (int)((1024u >> 16) | (64u << 16));  // dim1 hi | tile_dim0 = 64
  g1.s4 = 32;                                  // tile_dim1 = 32, tile_dim2 = 0
  g1.s5 = 256;                                 // tensor_dim0_stride lo32
  g1.s6 = 0;
  g1.s7 = 0;
  v4i_t gz4 = {0, 0, 0, 0};
  v8i_t gz8 = {0, 0, 0, 0, 0, 0, 0, 0};
  __builtin_amdgcn_tensor_load_to_lds(g0, g1, gz4, gz4, gz8, 0);
  __builtin_amdgcn_s_wait_tensorcnt(0);
  __syncthreads();
  sink[threadIdx.x] = (float)buf[threadIdx.x];
#else
  (void)src;
  if (threadIdx.x == 0) sink[0] = 0.0f;
#endif
}

// ---------------------------------------------------------------------------
// Persistent single-WGP LSTM recurrence.
// ---------------------------------------------------------------------------
__global__ __launch_bounds__(1024) void k_lstm_seq(
    const float* __restrict__ xg, const _Float16* __restrict__ Whh,
    const float* __restrict__ bhh, float* __restrict__ Hout, int T) {
  extern __shared__ char smem[];
  float* cst = (float*)smem;                     // 32*256 f32
  float* zs  = cst + 32 * 256;                   // 32*1024 f32
  _Float16* h16 = (_Float16*)(zs + 32 * 1024);   // 32*256 f16
  const int tid = threadIdx.x, lane = tid & 31, w = tid >> 5;
  const int mt = w & 1, wc = w >> 1;
  for (int i = tid; i < 32 * 256; i += 1024) { cst[i] = 0.0f; h16[i] = (_Float16)0.0f; }
  __syncthreads();
  for (int t = 0; t < T; ++t) {
    v8f zero = {};
    v8f cacc[4];
#pragma unroll
    for (int jj = 0; jj < 4; ++jj) cacc[jj] = zero;
    for (int ks = 0; ks < 8; ++ks) {
      v16h a = fragA(h16, 256, mt * 16, ks * 32, lane);
#pragma unroll
      for (int jj = 0; jj < 4; ++jj) {
        v16h b = fragB_nt(Whh, 256, (wc * 4 + jj) * 16, ks * 32, lane);
        cacc[jj] = wmma32(a, b, cacc[jj]);
      }
    }
    int hi = lane >> 4, nn = lane & 15;
#pragma unroll
    for (int jj = 0; jj < 4; ++jj)
#pragma unroll
      for (int r = 0; r < 8; ++r)
        zs[(mt * 16 + hi * 8 + r) * 1024 + (wc * 4 + jj) * 16 + nn] = cacc[jj][r];
    __syncthreads();
    for (int i = tid; i < 32 * 256; i += 1024) {
      int bb = i >> 8, u = i & 255;
      size_t row = ((size_t)bb * T + t) * 1024;
      float zi = xg[row + u]       + bhh[u]       + zs[bb * 1024 + u];
      float zf = xg[row + 256 + u] + bhh[256 + u] + zs[bb * 1024 + 256 + u];
      float zgv= xg[row + 512 + u] + bhh[512 + u] + zs[bb * 1024 + 512 + u];
      float zo = xg[row + 768 + u] + bhh[768 + u] + zs[bb * 1024 + 768 + u];
      float cn = sigf(zf) * cst[i] + sigf(zi) * tanhf(zgv);
      float hn = sigf(zo) * tanhf(cn);
      cst[i] = cn;
      h16[i] = (_Float16)hn;
      Hout[((size_t)t * 32 + bb) * 256 + u] = hn;
    }
    __syncthreads();
  }
}

// ---------------------------------------------------------------------------
// Persistent single-WGP match-LSTM.
// ---------------------------------------------------------------------------
__global__ __launch_bounds__(1024) void k_match_seq(
    const float* __restrict__ Hc, const float* __restrict__ Hq,
    const float* __restrict__ WQ, const float* __restrict__ qmask,
    const _Float16* __restrict__ attC, const _Float16* __restrict__ attM,
    const _Float16* __restrict__ mergeW, const float* __restrict__ mergeB,
    const _Float16* __restrict__ MWih, const _Float16* __restrict__ MWhh,
    const float* __restrict__ Mbih, const float* __restrict__ Mbhh,
    const float* __restrict__ scaleW, const float* __restrict__ scaleB,
    float* __restrict__ Hm, int T) {
  extern __shared__ char smem[];
  float* cm   = (float*)smem;          // 32*256
  float* s12  = cm + 8192;             // 32*256
  float* zg   = s12 + 8192;            // 32*1024
  float* satt = zg + 32768;            // 50*32
  float* aatt = satt + 1600;           // 50*32
  _Float16* hm16  = (_Float16*)(aatt + 1600);  // 32*256
  _Float16* hc16  = hm16 + 8192;               // 32*256
  _Float16* zin16 = hc16 + 8192;               // 32*512
  _Float16* z16   = zin16 + 16384;             // 32*256
  const int tid = threadIdx.x, lane = tid & 31, w = tid >> 5;
  for (int i = tid; i < 8192; i += 1024) { cm[i] = 0.0f; hm16[i] = (_Float16)0.0f; }
  __syncthreads();
  for (int t = 0; t < T; ++t) {
    for (int i = tid; i < 8192; i += 1024) {
      float v = Hc[((size_t)t * 32) * 256 + i];
      hc16[i] = (_Float16)v;
      zin16[(i >> 8) * 512 + (i & 255)] = (_Float16)v;
    }
    __syncthreads();
    { // S = hc@attC^T + hm@attM^T
      int mt = w & 1, nt = w >> 1;
      v8f c = {};
      for (int ks = 0; ks < 8; ++ks)
        c = wmma32(fragA(hc16, 256, mt * 16, ks * 32, lane),
                   fragB_nt(attC, 256, nt * 16, ks * 32, lane), c);
      for (int ks = 0; ks < 8; ++ks)
        c = wmma32(fragA(hm16, 256, mt * 16, ks * 32, lane),
                   fragB_nt(attM, 256, nt * 16, ks * 32, lane), c);
      int hi = lane >> 4, nn = lane & 15;
#pragma unroll
      for (int r = 0; r < 8; ++r)
        s12[(mt * 16 + hi * 8 + r) * 256 + nt * 16 + nn] = c[r];
    }
    __syncthreads();
    for (int rr = w; rr < 1600; rr += 32) {
      int bb = rr & 31;
      float part = 0.0f;
      for (int h = lane; h < 256; h += 32)
        part += tanhf(WQ[(size_t)rr * 256 + h] + s12[bb * 256 + h]) * scaleW[h];
      for (int o = 16; o > 0; o >>= 1) part += __shfl_xor(part, o, 32);
      if (lane == 0) satt[rr] = part + scaleB[0];
    }
    __syncthreads();
    { // masked softmax over tq, wave w = batch b
      int bb = w;
      int t0 = lane, t1 = lane + 32;
      float m0 = (t0 < 50) ? qmask[bb * 50 + t0] : 1.0f;
      float m1 = (t1 < 50) ? qmask[bb * 50 + t1] : 1.0f;
      float x0 = (t0 < 50) ? (satt[t0 * 32 + bb] * (1.f - m0) + m0 * (-1e30f)) : -3.4e38f;
      float x1 = (t1 < 50) ? (satt[t1 * 32 + bb] * (1.f - m1) + m1 * (-1e30f)) : -3.4e38f;
      float mx = fmaxf(x0, x1);
      for (int o = 16; o > 0; o >>= 1) mx = fmaxf(mx, __shfl_xor(mx, o, 32));
      float e0 = (t0 < 50) ? __expf(x0 - mx) : 0.0f;
      float e1 = (t1 < 50) ? __expf(x1 - mx) : 0.0f;
      float sm = e0 + e1;
      for (int o = 16; o > 0; o >>= 1) sm += __shfl_xor(sm, o, 32);
      float inv = 1.0f / sm;
      if (t0 < 50) aatt[t0 * 32 + bb] = e0 * inv;
      if (t1 < 50) aatt[t1 * 32 + bb] = e1 * inv;
    }
    __syncthreads();
    for (int i = tid; i < 8192; i += 1024) {
      int bb = i >> 8, u = i & 255;
      float acc = 0.0f;
      for (int tq = 0; tq < 50; ++tq)
        acc += aatt[tq * 32 + bb] * Hq[((size_t)tq * 32 + bb) * 256 + u];
      zin16[bb * 512 + 256 + u] = (_Float16)acc;
    }
    __syncthreads();
    { // z = [hc,ctx]@merge^T + merge_b
      int mt = w & 1, nt = w >> 1;
      v8f c = {};
      for (int ks = 0; ks < 16; ++ks)
        c = wmma32(fragA(zin16, 512, mt * 16, ks * 32, lane),
                   fragB_nt(mergeW, 512, nt * 16, ks * 32, lane), c);
      int hi = lane >> 4, nn = lane & 15;
#pragma unroll
      for (int r = 0; r < 8; ++r) {
        int m = mt * 16 + hi * 8 + r, n = nt * 16 + nn;
        z16[m * 256 + n] = (_Float16)(c[r] + mergeB[n]);
      }
    }
    __syncthreads();
    { // gates = z@MWih^T + hm@MWhh^T
      int mt = w & 1, wc = w >> 1;
      v8f zero = {};
      v8f cacc[4];
#pragma unroll
      for (int jj = 0; jj < 4; ++jj) cacc[jj] = zero;
      for (int ks = 0; ks < 8; ++ks) {
        v16h a = fragA(z16, 256, mt * 16, ks * 32, lane);
#pragma unroll
        for (int jj = 0; jj < 4; ++jj)
          cacc[jj] = wmma32(a, fragB_nt(MWih, 256, (wc * 4 + jj) * 16, ks * 32, lane), cacc[jj]);
      }
      for (int ks = 0; ks < 8; ++ks) {
        v16h a = fragA(hm16, 256, mt * 16, ks * 32, lane);
#pragma unroll
        for (int jj = 0; jj < 4; ++jj)
          cacc[jj] = wmma32(a, fragB_nt(MWhh, 256, (wc * 4 + jj) * 16, ks * 32, lane), cacc[jj]);
      }
      int hi = lane >> 4, nn = lane & 15;
#pragma unroll
      for (int jj = 0; jj < 4; ++jj)
#pragma unroll
        for (int r = 0; r < 8; ++r)
          zg[(mt * 16 + hi * 8 + r) * 1024 + (wc * 4 + jj) * 16 + nn] = cacc[jj][r];
    }
    __syncthreads();
    for (int i = tid; i < 8192; i += 1024) {
      int bb = i >> 8, u = i & 255;
      float zi = zg[bb * 1024 + u]       + Mbih[u]       + Mbhh[u];
      float zf = zg[bb * 1024 + 256 + u] + Mbih[256 + u] + Mbhh[256 + u];
      float zgv= zg[bb * 1024 + 512 + u] + Mbih[512 + u] + Mbhh[512 + u];
      float zo = zg[bb * 1024 + 768 + u] + Mbih[768 + u] + Mbhh[768 + u];
      float cn = sigf(zf) * cm[i] + sigf(zi) * tanhf(zgv);
      float hn = sigf(zo) * tanhf(cn);
      cm[i] = cn;
      hm16[i] = (_Float16)hn;
      Hm[((size_t)t * 32 + bb) * 256 + u] = hn;
    }
    __syncthreads();
  }
}

// ---------------------------------------------------------------------------
// Small supporting kernels
// ---------------------------------------------------------------------------
__global__ void k_mask(const int* __restrict__ wid, float* __restrict__ m, int n) {
  int i = blockIdx.x * blockDim.x + threadIdx.x;
  if (i < n) m[i] = (wid[i] == 0) ? 1.0f : 0.0f;
}
__global__ void k_pe(float* __restrict__ pe) {
  int i = blockIdx.x * blockDim.x + threadIdx.x;
  if (i >= 96 * 400) return;
  int d = i / 400, l = i % 400;
  double freq, phase;
  if ((d & 1) == 0) { freq = pow(10000.0, -(double)d / 96.0); phase = 0.0; }
  else { freq = -pow(10000.0, (1.0 - d) / 96.0); phase = 1.5707963267948966; }
  pe[i] = (float)sin((double)l * freq + phase);
}
__global__ void k_f32h(_Float16* __restrict__ y, const float* __restrict__ x, int n) {
  int i = blockIdx.x * blockDim.x + threadIdx.x;
  if (i < n) y[i] = (_Float16)x[i];
}
__global__ void k_char_gather(float* __restrict__ out, const float* __restrict__ cmat,
                              const int* __restrict__ cid, int L, int n) {
  int i = blockIdx.x * blockDim.x + threadIdx.x;
  if (i >= n) return;
  int j = i & 15, v = i >> 4;
  int l = v % L, u = v / L;
  int c = u & 63, bb = u >> 6;
  out[i] = cmat[(size_t)cid[((size_t)bb * L + l) * 16 + j] * 64 + c];
}
__global__ void k_char_dw(float* __restrict__ y, const float* __restrict__ x,
                          const float* __restrict__ w5, const float* __restrict__ bias,
                          int L, int n) {
  int i = blockIdx.x * blockDim.x + threadIdx.x;
  if (i >= n) return;
  int j = i & 15, v = i >> 4;
  int l = v % L, u = v / L;
  int c = u & 63, bb = u >> 6;
  float acc = bias[c];
  for (int du = 0; du < 5; ++du) {
    int ll = l + du - 2;
    if (ll < 0 || ll >= L) continue;
    for (int dv = 0; dv < 5; ++dv) {
      int jj = j + dv - 2;
      if (jj < 0 || jj >= 16) continue;
      acc += w5[c * 25 + du * 5 + dv] * x[(((size_t)bb * 64 + c) * L + ll) * 16 + jj];
    }
  }
  y[i] = acc;
}
__global__ void k_char_pw(float* __restrict__ chm, const float* __restrict__ y,
                          const float* __restrict__ pw, const float* __restrict__ pb,
                          int L, int n) {
  int i = blockIdx.x * blockDim.x + threadIdx.x;
  if (i >= n) return;
  int l = i % L, u = i / L;
  int c = u & 63, bb = u >> 6;
  float best = 0.0f;
  for (int j = 0; j < 16; ++j) {
    float acc = pb[c];
    for (int ci = 0; ci < 64; ++ci)
      acc += pw[c * 64 + ci] * y[(((size_t)bb * 64 + ci) * L + l) * 16 + j];
    best = fmaxf(best, fmaxf(acc, 0.0f));
  }
  chm[i] = best;
}
__global__ void k_embed_concat(float* __restrict__ E, const float* __restrict__ chm,
                               const float* __restrict__ wmat, const int* __restrict__ wid,
                               int L, int n) {
  int i = blockIdx.x * blockDim.x + threadIdx.x;
  if (i >= n) return;
  int cf = i % 364, tok = i / 364;
  int bb = tok / L, l = tok % L;
  if (cf < 64) E[i] = chm[((size_t)bb * 64 + cf) * L + l];
  else E[i] = wmat[(size_t)wid[tok] * 300 + (cf - 64)];
}
__global__ void k_hw_mix(float* __restrict__ E, const float* __restrict__ G,
                         const float* __restrict__ NL, int n) {
  int i = blockIdx.x * blockDim.x + threadIdx.x;
  if (i < n) E[i] = G[i] * NL[i] + (1.0f - G[i]) * E[i];
}
__global__ void k_add(float* __restrict__ d, const float* __restrict__ a,
                      const float* __restrict__ b, int n) {
  int i = blockIdx.x * blockDim.x + threadIdx.x;
  if (i < n) d[i] = a[i] + b[i];
}
__global__ void k_add_pe(float* __restrict__ x, const float* __restrict__ pe, int n) {
  int i = blockIdx.x * blockDim.x + threadIdx.x;
  if (i < n) x[i] += pe[i % (96 * 400)];
}
// LayerNorm over (DM,LC) per batch; dual f32/f16 output (f16 feeds WMMA GEMMs)
__global__ __launch_bounds__(256) void k_ln(float* __restrict__ out,
                                            _Float16* __restrict__ outh,
                                            const float* __restrict__ x,
                                            const float* __restrict__ g,
                                            const float* __restrict__ beta) {
  __shared__ float rs[256], rq[256];
  const int tid = threadIdx.x;
  const float* xb = x + (size_t)blockIdx.x * 38400;
  float s = 0.0f, q = 0.0f;
  for (int i = tid; i < 38400; i += 256) { float v = xb[i]; s += v; q += v * v; }
  rs[tid] = s; rq[tid] = q; __syncthreads();
  for (int o = 128; o > 0; o >>= 1) {
    if (tid < o) { rs[tid] += rs[tid + o]; rq[tid] += rq[tid + o]; }
    __syncthreads();
  }
  float mu = rs[0] / 38400.0f;
  float var = rq[0] / 38400.0f - mu * mu;
  float rinv = rsqrtf(var + 1e-5f);
  float* ob = out + (size_t)blockIdx.x * 38400;
  _Float16* oh = outh + (size_t)blockIdx.x * 38400;
  for (int i = tid; i < 38400; i += 256) {
    float v = (xb[i] - mu) * rinv * g[i] + beta[i];
    ob[i] = v;
    oh[i] = (_Float16)v;
  }
}
// depthwise 5-tap conv; output f16 (feeds pointwise WMMA GEMM only)
__global__ void k_dwconv(_Float16* __restrict__ y, const float* __restrict__ x,
                         const float* __restrict__ w5, const float* __restrict__ bias,
                         int n) {
  int i = blockIdx.x * blockDim.x + threadIdx.x;
  if (i >= n) return;
  int l = i % 400, d = (i / 400) % 96;
  int base = i - l;
  float acc = bias[d];
#pragma unroll
  for (int k = 0; k < 5; ++k) {
    int ll = l + k - 2;
    if (ll >= 0 && ll < 400) acc += w5[d * 5 + k] * x[base + ll];
  }
  y[i] = (_Float16)acc;
}
__global__ __launch_bounds__(128) void k_attn(
    _Float16* __restrict__ AO, const float* __restrict__ Q,
    const float* __restrict__ Kt, const float* __restrict__ V,
    const float* __restrict__ cmask) {
  __shared__ float probs[400];
  __shared__ float red[128];
  __shared__ float opart[12 * 128];
  const int l = blockIdx.x, h = blockIdx.y, bb = blockIdx.z;
  const int tid = threadIdx.x;
  const size_t base = ((size_t)bb * 96 + h * 12) * 400;
  float qd[12];
#pragma unroll
  for (int d = 0; d < 12; ++d) qd[d] = Q[base + d * 400 + l];
  float lmax = -3.4e38f;
  for (int m = tid; m < 400; m += 128) {
    float s = 0.0f;
#pragma unroll
    for (int d = 0; d < 12; ++d) s += qd[d] * Kt[base + d * 400 + m];
    s *= 0.2886751345948129f;  // 1/sqrt(12)
    float mk = cmask[bb * 400 + m];
    s = s * (1.0f - mk) + mk * (-1e30f);
    probs[m] = s;
    lmax = fmaxf(lmax, s);
  }
  red[tid] = lmax; __syncthreads();
  for (int o = 64; o > 0; o >>= 1) { if (tid < o) red[tid] = fmaxf(red[tid], red[tid + o]); __syncthreads(); }
  float mx = red[0];
  __syncthreads();
  float lsum = 0.0f;
  for (int m = tid; m < 400; m += 128) { float e = __expf(probs[m] - mx); probs[m] = e; lsum += e; }
  red[tid] = lsum; __syncthreads();
  for (int o = 64; o > 0; o >>= 1) { if (tid < o) red[tid] += red[tid + o]; __syncthreads(); }
  float inv = 1.0f / red[0];
  float acc[12];
#pragma unroll
  for (int d = 0; d < 12; ++d) acc[d] = 0.0f;
  for (int m = tid; m < 400; m += 128) {
    float pv = probs[m] * inv;
#pragma unroll
    for (int d = 0; d < 12; ++d) acc[d] += pv * V[base + d * 400 + m];
  }
#pragma unroll
  for (int d = 0; d < 12; ++d) opart[d * 128 + tid] = acc[d];
  __syncthreads();
  if (tid < 12) {
    float s = 0.0f;
    for (int i = 0; i < 128; ++i) s += opart[tid * 128 + i];
    AO[base + tid * 400 + l] = (_Float16)s;
  }
}
__global__ void k_repack(float* __restrict__ M, const float* __restrict__ rows, int n) {
  int i = blockIdx.x * blockDim.x + threadIdx.x;
  if (i >= n) return;
  int bb = i / (96 * 400), r = i % (96 * 400);
  int d = r / 400, t = r % 400;
  M[i] = rows[((size_t)t * 32 + bb) * 96 + d];
}
__global__ void k_ptr(float* __restrict__ Y, const float* __restrict__ Ma,
                      const float* __restrict__ Mb, const float* __restrict__ w, int n) {
  int i = blockIdx.x * blockDim.x + threadIdx.x;
  if (i >= n) return;
  int bb = i / 400, l = i % 400;
  float acc = 0.0f;
  for (int d = 0; d < 96; ++d) {
    acc += w[d] * Ma[((size_t)bb * 96 + d) * 400 + l];
    acc += w[96 + d] * Mb[((size_t)bb * 96 + d) * 400 + l];
  }
  Y[i] = acc;
}
__global__ __launch_bounds__(256) void k_logsoftmax(float* __restrict__ out,
                                                    const float* __restrict__ Y,
                                                    const float* __restrict__ cmask) {
  __shared__ float red[256];
  int bb = blockIdx.x, tid = threadIdx.x;
  float lmax = -3.4e38f;
  for (int l = tid; l < 400; l += 256) {
    float mk = cmask[bb * 400 + l];
    float v = Y[bb * 400 + l] * (1.0f - mk) + mk * (-1e30f);
    lmax = fmaxf(lmax, v);
  }
  red[tid] = lmax; __syncthreads();
  for (int o = 128; o > 0; o >>= 1) { if (tid < o) red[tid] = fmaxf(red[tid], red[tid + o]); __syncthreads(); }
  float mx = red[0];
  __syncthreads();
  float ls = 0.0f;
  for (int l = tid; l < 400; l += 256) {
    float mk = cmask[bb * 400 + l];
    float v = Y[bb * 400 + l] * (1.0f - mk) + mk * (-1e30f);
    ls += __expf(v - mx);
  }
  red[tid] = ls; __syncthreads();
  for (int o = 128; o > 0; o >>= 1) { if (tid < o) red[tid] += red[tid + o]; __syncthreads(); }
  float lse = logf(red[0]);
  for (int l = tid; l < 400; l += 256) {
    float mk = cmask[bb * 400 + l];
    float v = Y[bb * 400 + l] * (1.0f - mk) + mk * (-1e30f);
    out[bb * 400 + l] = v - mx - lse;
  }
}

// ---------------------------------------------------------------------------
// Host orchestration
// ---------------------------------------------------------------------------
extern "C" void kernel_launch(void* const* d_in, const int* in_sizes, int n_in,
                              void* d_out, int out_size, void* d_ws, size_t ws_size,
                              hipStream_t stream) {
  (void)in_sizes; (void)n_in; (void)out_size; (void)ws_size;
  auto F = [&](int i) { return (const float*)d_in[i]; };
  auto Iv = [&](int i) { return (const int*)d_in[i]; };
  auto cdiv = [](int a, int b) { return (a + b - 1) / b; };

  const float* word_mat = F(0);
  const float* char_mat = F(1);
  const float* emb_dw_w = F(2); const float* emb_dw_b = F(3);
  const float* emb_pw_w = F(4); const float* emb_pw_b = F(5);
  const float* hw_lin_w[2]  = {F(6), F(10)};
  const float* hw_lin_b[2]  = {F(7), F(11)};
  const float* hw_gate_w[2] = {F(8), F(12)};
  const float* hw_gate_b[2] = {F(9), F(13)};
  const float* C_Wih = F(14); const float* C_Whh = F(15);
  const float* C_bih = F(16); const float* C_bhh = F(17);
  const float* Q_Wih = F(18); const float* Q_Whh = F(19);
  const float* Q_bih = F(20); const float* Q_bhh = F(21);
  const float* M_Wih = F(22); const float* M_Whh = F(23);
  const float* M_bih = F(24); const float* M_bhh = F(25);
  const float* att_C = F(26); const float* att_Q = F(27); const float* att_M = F(28);
  const float* scale_w = F(29); const float* scale_b = F(30);
  const float* merge_w = F(31); const float* merge_b = F(32);
  const float* my_w = F(33); const float* my_b = F(34);
  const float* eb_dw_w[2] = {F(35), F(39)};
  const float* eb_dw_b[2] = {F(36), F(40)};
  const float* eb_pw_w[2] = {F(37), F(41)};
  const float* eb_pw_b[2] = {F(38), F(42)};
  const float* mha_q_w = F(43); const float* mha_q_b = F(44);
  const float* mha_k_w = F(45); const float* mha_k_b = F(46);
  const float* mha_v_w = F(47); const float* mha_v_b = F(48);
  const float* mha_o_w = F(49); const float* mha_o_b = F(50);
  const float* eb_fc_w = F(51); const float* eb_fc_b = F(52);
  const float* ln_g[4] = {F(53), F(55), F(57), F(59)};
  const float* ln_b[4] = {F(54), F(56), F(58), F(60)};
  const float* ptr_w1 = F(61); const float* ptr_w2 = F(62);
  const int* Cwid = Iv(63); const int* Ccid = Iv(64);
  const int* Qwid = Iv(65); const int* Qcid = Iv(66);

  char* wsp = (char*)d_ws;
  size_t off = 0;
  auto alloc = [&](size_t bytes) {
    void* p = wsp + off;
    off += (bytes + 255) & ~(size_t)255;
    return p;
  };
  float* BSA = (float*)alloc(52428800);
  float* BSB = (float*)alloc(52428800);
  float* E_c = (float*)alloc((size_t)12800 * 364 * 4);
  float* E_q = (float*)alloc((size_t)1600 * 364 * 4);
  float* Hc  = (float*)alloc((size_t)400 * 32 * 256 * 4);
  float* Hq  = (float*)alloc((size_t)50 * 32 * 256 * 4);
  float* WQb = (float*)alloc((size_t)50 * 32 * 256 * 4);
  float* Hm  = (float*)alloc((size_t)400 * 32 * 256 * 4);
  size_t nBDL = (size_t)32 * 96 * 400;
  float* M1 = (float*)alloc(nBDL * 4);
  float* M2 = (float*)alloc(nBDL * 4);
  float* M3 = (float*)alloc(nBDL * 4);
  float* T1 = (float*)alloc(nBDL * 4);
  float* T2 = (float*)alloc(nBDL * 4);
  float* Qb = (float*)alloc(nBDL * 4);
  float* Kb = (float*)alloc(nBDL * 4);
  float* Vb = (float*)alloc(nBDL * 4);
  float* peB = (float*)alloc(96 * 400 * 4);
  float* cmaskB = (float*)alloc(32 * 400 * 4);
  float* qmaskB = (float*)alloc(32 * 50 * 4);
  float* Y1 = (float*)alloc(32 * 400 * 4);
  float* Y2 = (float*)alloc(32 * 400 * 4);
  float* sinkB = (float*)alloc(256 * 4);
  _Float16* CWhh_h = (_Float16*)alloc(1024 * 256 * 2);
  _Float16* QWhh_h = (_Float16*)alloc(1024 * 256 * 2);
  _Float16* MWhh_h = (_Float16*)alloc(1024 * 256 * 2);
  _Float16* MWih_h = (_Float16*)alloc(1024 * 256 * 2);
  _Float16* attC_h = (_Float16*)alloc(256 * 256 * 2);
  _Float16* attM_h = (_Float16*)alloc(256 * 256 * 2);
  _Float16* merge_h = (_Float16*)alloc(256 * 512 * 2);
  _Float16* hwgH[2] = {(_Float16*)alloc(364 * 364 * 2), (_Float16*)alloc(364 * 364 * 2)};
  _Float16* hwlH[2] = {(_Float16*)alloc(364 * 364 * 2), (_Float16*)alloc(364 * 364 * 2)};
  _Float16* CWihH = (_Float16*)alloc(1024 * 364 * 2);
  _Float16* QWihH = (_Float16*)alloc(1024 * 364 * 2);
  _Float16* attQH = (_Float16*)alloc(256 * 256 * 2);
  _Float16* mywH  = (_Float16*)alloc(96 * 256 * 2);
  _Float16* pwH[2] = {(_Float16*)alloc(96 * 96 * 2), (_Float16*)alloc(96 * 96 * 2)};
  _Float16* qH = (_Float16*)alloc(96 * 96 * 2);
  _Float16* kH = (_Float16*)alloc(96 * 96 * 2);
  _Float16* vH = (_Float16*)alloc(96 * 96 * 2);
  _Float16* oH = (_Float16*)alloc(96 * 96 * 2);
  _Float16* fcH = (_Float16*)alloc(96 * 96 * 2);
  _Float16* AH  = (_Float16*)alloc((size_t)12800 * 364 * 2);  // activation f16 scratch
  _Float16* T1H = (_Float16*)alloc(nBDL * 2);
  _Float16* DWH = (_Float16*)alloc(nBDL * 2);
  _Float16* AOH = (_Float16*)alloc(nBDL * 2);

  auto toh = [&](_Float16* d, const float* s, int n) {
    k_f32h<<<cdiv(n, 256), 256, 0, stream>>>(d, s, n);
  };
  auto gemm_nt = [&](const _Float16* A, const _Float16* W, const float* bias, float* C,
                     int M, int N, int K, int act) {
    dim3 g(cdiv(N, 64), cdiv(M, 32));
    k_gemm_nt<<<g, 256, 0, stream>>>(A, W, bias, C, M, N, K, act);
  };
  auto gemm_nn = [&](float* C, const _Float16* W, const _Float16* X, const float* bias,
                     int M, int N, int K, int act) {
    dim3 g(cdiv(N, 64), cdiv(M, 32), 32);
    k_gemm_nn<<<g, 256, 0, stream>>>(W, X, bias, C, M, N, K, act);
  };

  // masks, pe, weight f16 staging
  k_mask<<<cdiv(32 * 400, 256), 256, 0, stream>>>(Cwid, cmaskB, 32 * 400);
  k_mask<<<cdiv(32 * 50, 256), 256, 0, stream>>>(Qwid, qmaskB, 32 * 50);
  k_pe<<<cdiv(96 * 400, 256), 256, 0, stream>>>(peB);
  toh(CWhh_h, C_Whh, 1024 * 256); toh(QWhh_h, Q_Whh, 1024 * 256);
  toh(MWhh_h, M_Whh, 1024 * 256); toh(MWih_h, M_Wih, 1024 * 256);
  toh(attC_h, att_C, 256 * 256); toh(attM_h, att_M, 256 * 256);
  toh(merge_h, merge_w, 256 * 512);
  for (int i = 0; i < 2; ++i) {
    toh(hwgH[i], hw_gate_w[i], 364 * 364);
    toh(hwlH[i], hw_lin_w[i], 364 * 364);
    toh(pwH[i], eb_pw_w[i], 96 * 96);
  }
  toh(CWihH, C_Wih, 1024 * 364); toh(QWihH, Q_Wih, 1024 * 364);
  toh(attQH, att_Q, 256 * 256); toh(mywH, my_w, 96 * 256);
  toh(qH, mha_q_w, 96 * 96); toh(kH, mha_k_w, 96 * 96);
  toh(vH, mha_v_w, 96 * 96); toh(oH, mha_o_w, 96 * 96);
  toh(fcH, eb_fc_w, 96 * 96);

  // embedding (char conv + concat)
  auto run_embed = [&](const int* wid, const int* cid, float* E, int L) {
    int ng = 32 * 64 * L * 16;
    k_char_gather<<<cdiv(ng, 256), 256, 0, stream>>>(BSA, char_mat, cid, L, ng);
    k_char_dw<<<cdiv(ng, 256), 256, 0, stream>>>(BSB, BSA, emb_dw_w, emb_dw_b, L, ng);
    int np = 32 * 64 * L;
    k_char_pw<<<cdiv(np, 256), 256, 0, stream>>>(BSA, BSB, emb_pw_w, emb_pw_b, L, np);
    int ne = 32 * L * 364;
    k_embed_concat<<<cdiv(ne, 256), 256, 0, stream>>>(E, BSA, word_mat, wid, L, ne);
  };
  run_embed(Cwid, Ccid, E_c, 400);
  run_embed(Qwid, Qcid, E_q, 50);

  // highway (2 layers): E is mutated per layer, so re-stage f16 each layer
  auto run_highway = [&](float* E, int rows) {
    for (int i = 0; i < 2; ++i) {
      toh(AH, E, rows * 364);
      gemm_nt(AH, hwgH[i], hw_gate_b[i], BSA, rows, 364, 364, 2);
      gemm_nt(AH, hwlH[i], hw_lin_b[i], BSB, rows, 364, 364, 1);
      int n = rows * 364;
      k_hw_mix<<<cdiv(n, 256), 256, 0, stream>>>(E, BSA, BSB, n);
    }
  };
  run_highway(E_c, 12800);
  run_highway(E_q, 1600);

  // LSTM scans
  k_tdm_warm<<<1, 32, 0, stream>>>(CWhh_h, sinkB);  // TDM path probe / L2 warm
  toh(AH, E_c, 12800 * 364);
  gemm_nt(AH, CWihH, C_bih, BSA, 12800, 1024, 364, 0);
  k_lstm_seq<<<1, 1024, 180224, stream>>>(BSA, CWhh_h, C_bhh, Hc, 400);
  toh(AH, E_q, 1600 * 364);
  gemm_nt(AH, QWihH, Q_bih, BSB, 1600, 1024, 364, 0);
  k_lstm_seq<<<1, 1024, 180224, stream>>>(BSB, QWhh_h, Q_bhh, Hq, 50);

  // match-LSTM
  toh(AH, Hq, 1600 * 256);
  gemm_nt(AH, attQH, nullptr, WQb, 1600, 256, 256, 0);
  k_match_seq<<<1, 1024, 291328, stream>>>(Hc, Hq, WQb, qmaskB, attC_h, attM_h,
                                           merge_h, merge_b, MWih_h, MWhh_h,
                                           M_bih, M_bhh, scale_w, scale_b, Hm, 400);

  // projection to model dim + repack to [B, DM, LC]
  toh(AH, Hm, 12800 * 256);
  gemm_nt(AH, mywH, my_b, BSB, 12800, 96, 256, 0);
  k_repack<<<cdiv((int)nBDL, 256), 256, 0, stream>>>(M1, BSB, (int)nBDL);

  // encoder block (shared weights, 21 applications)
  auto enc_block = [&](float* X) {
    int n = (int)nBDL;
    k_add_pe<<<cdiv(n, 256), 256, 0, stream>>>(X, peB, n);
    k_ln<<<32, 256, 0, stream>>>(T1, T1H, X, ln_g[0], ln_b[0]);
    for (int i = 0; i < 2; ++i) {
      k_dwconv<<<cdiv(n, 256), 256, 0, stream>>>(DWH, T1, eb_dw_w[i], eb_dw_b[i], n);
      gemm_nn(T2, pwH[i], DWH, eb_pw_b[i], 96, 400, 96, 1);
      k_add<<<cdiv(n, 256), 256, 0, stream>>>(X, X, T2, n);
      k_ln<<<32, 256, 0, stream>>>(T1, T1H, X, ln_g[1 + i], ln_b[1 + i]);
    }
    gemm_nn(Qb, qH, T1H, mha_q_b, 96, 400, 96, 0);
    gemm_nn(Kb, kH, T1H, mha_k_b, 96, 400, 96, 0);
    gemm_nn(Vb, vH, T1H, mha_v_b, 96, 400, 96, 0);
    k_attn<<<dim3(400, 8, 32), 128, 0, stream>>>(AOH, Qb, Kb, Vb, cmaskB);
    gemm_nn(T2, oH, AOH, mha_o_b, 96, 400, 96, 0);
    k_add<<<cdiv(n, 256), 256, 0, stream>>>(X, X, T2, n);
    k_ln<<<32, 256, 0, stream>>>(T1, T1H, X, ln_g[3], ln_b[3]);
    gemm_nn(T2, fcH, T1H, eb_fc_b, 96, 400, 96, 1);
    k_add<<<cdiv(n, 256), 256, 0, stream>>>(X, X, T2, n);
  };
  for (int i = 0; i < 7; ++i) enc_block(M1);
  (void)hipMemcpyAsync(M2, M1, nBDL * 4, hipMemcpyDeviceToDevice, stream);
  for (int i = 0; i < 7; ++i) enc_block(M2);
  (void)hipMemcpyAsync(M3, M2, nBDL * 4, hipMemcpyDeviceToDevice, stream);
  for (int i = 0; i < 7; ++i) enc_block(M3);

  // pointers + masked log-softmax -> d_out [2, B, LC]
  float* out = (float*)d_out;
  k_ptr<<<cdiv(32 * 400, 256), 256, 0, stream>>>(Y1, M1, M2, ptr_w1, 32 * 400);
  k_ptr<<<cdiv(32 * 400, 256), 256, 0, stream>>>(Y2, M1, M3, ptr_w2, 32 * 400);
  k_logsoftmax<<<32, 256, 0, stream>>>(out, Y1, cmaskB);
  k_logsoftmax<<<32, 256, 0, stream>>>(out + 12800, Y2, cmaskB);
}
